// RelativeMultiheadAttentionMechanism_2473901162553
// MI455X (gfx1250) — compile-verified, matrix operand order checked by reading, not run on hardware
//
#include <hip/hip_runtime.h>
#include <hip/hip_bf16.h>
#include <stdint.h>

// ---------------------------------------------------------------------------
// Transformer-XL relative multi-head attention for MI455X (gfx1250, wave32).
// GEMMs use v_wmma_f32_16x16x32_bf16; deep-K GEMMs use 128x128 block tiles
// with the shared B panel staged through LDS by double-buffered
// GLOBAL_LOAD_ASYNC_TO_LDS_B128 (ASYNCcnt + split barriers).
// ---------------------------------------------------------------------------

typedef __bf16 bf16_t;
typedef __attribute__((ext_vector_type(16))) __bf16 v16bf;
typedef __attribute__((ext_vector_type(8)))  float  v8f;

#define DEVINL __device__ __forceinline__

constexpr int B_    = 2;
constexpr int QLEN_ = 1024;
constexpr int MLEN_ = 1024;
constexpr int C_    = 1024;
constexpr int H_    = 16;
constexpr int DK_   = 64;
constexpr int KLEN_ = 2048;   // QLEN + MLEN

constexpr int ROWB  = 40;     // LDS row pitch in bf16 elems (80B: 64B data + 16B pad)

// ---- fragment loads -------------------------------------------------------
// 16-bit A/B fragment layout (ISA 7.12.2): lanes 0-15 hold row M=L with
// K = [0..7] and [16..23]; lanes 16-31 hold the same row with K = [8..15]
// and [24..31]: two 16-byte chunks at +hs*8 and +16+hs*8 elements.
DEVINL v16bf load_frag_bf(const bf16_t* p, int hs) {
  union { uint4 u[2]; v16bf v; } r;
  r.u[0] = *(const uint4*)(p + hs * 8);
  r.u[1] = *(const uint4*)(p + 16 + hs * 8);
  return r.v;
}

DEVINL v16bf load_frag_f32(const float* p, int hs) {
  const float4 x0 = *(const float4*)(p + hs * 8);
  const float4 x1 = *(const float4*)(p + hs * 8 + 4);
  const float4 y0 = *(const float4*)(p + 16 + hs * 8);
  const float4 y1 = *(const float4*)(p + 16 + hs * 8 + 4);
  v16bf r;
  r[0]  = (bf16_t)x0.x; r[1]  = (bf16_t)x0.y; r[2]  = (bf16_t)x0.z; r[3]  = (bf16_t)x0.w;
  r[4]  = (bf16_t)x1.x; r[5]  = (bf16_t)x1.y; r[6]  = (bf16_t)x1.z; r[7]  = (bf16_t)x1.w;
  r[8]  = (bf16_t)y0.x; r[9]  = (bf16_t)y0.y; r[10] = (bf16_t)y0.z; r[11] = (bf16_t)y0.w;
  r[12] = (bf16_t)y1.x; r[13] = (bf16_t)y1.y; r[14] = (bf16_t)y1.z; r[15] = (bf16_t)y1.w;
  return r;
}

DEVINL v8f wmma_bf16(v16bf a, v16bf b, v8f c) {
  return __builtin_amdgcn_wmma_f32_16x16x32_bf16(
      false, a, false, b, (short)0, c, false, false);
}

// ---- CDNA5 async global->LDS copy (ASYNCcnt-tracked) ----------------------
// Flat LDS aperture: low 32 bits of a __shared__ flat pointer are the LDS
// byte address, which is what the VDST operand of the async load expects.
DEVINL void async_ld_b128(const bf16_t* gsrc, void* ldsDst) {
  int loff = (int)(uintptr_t)ldsDst;
  asm volatile("global_load_async_to_lds_b128 %0, %1, off"
               :: "v"(loff), "v"((const void*)gsrc)
               : "memory");
}
DEVINL void wait_async0() {
  asm volatile("s_wait_asynccnt 0x0" ::: "memory");
}

// ---- fp32 -> bf16 conversion ---------------------------------------------
__global__ void cvt_bf16(const float* __restrict__ src, bf16_t* __restrict__ dst, int n) {
  int i = blockIdx.x * blockDim.x + threadIdx.x;
  const int stride = gridDim.x * blockDim.x;
  for (; i < n; i += stride) dst[i] = (bf16_t)src[i];
}

// ---- projection GEMM: OUT[M x C] = A[M x C] @ W^T -------------------------
// Block: 128 rows x 128-col panel. 8 waves, each 16 rows across 8 n-subtiles.
// W k-slice (128 x 32) staged in LDS, double-buffered async (2 b128/thread).
// blockIdx.x = bm * 8 + bn; grid exact (no early returns; kernel barriers).
// qremap: A row r -> key row (r/Q)*KLEN + MLEN + (r%Q)   (key[:, -Q:])
// vtrans: store transposed per head: vT[((b*H+h)*DK + d)*KLEN + j]
__global__ void __launch_bounds__(256)
proj_gemm(const bf16_t* __restrict__ A, const bf16_t* __restrict__ W,
          bf16_t* __restrict__ out0, bf16_t* __restrict__ out1,
          const float* __restrict__ bias0, const float* __restrict__ bias1,
          int qremap, int vtrans)
{
  __shared__ bf16_t Bsm[2][128 * ROWB];        // 2 x 10240 B
  const int tid  = threadIdx.x;
  const int lane = tid & 31;
  const int wv   = tid >> 5;
  const int bm   = blockIdx.x >> 3;
  const int bn   = blockIdx.x & 7;
  const int m0   = bm * 128 + wv * 16;
  const int n0   = bn * 128;
  const int r    = lane & 15, hs = (lane >> 4) & 1;

  // staging role: thread loads 32B: row = tid/2, chunks (tid&1)*2 .. +1
  const int srow = tid >> 1, sc = (tid & 1) * 16;   // element offset within row
  const bf16_t* gsrc = W + (size_t)(n0 + srow) * C_ + sc;
  bf16_t* l0 = &Bsm[0][srow * ROWB + sc];
  bf16_t* l1 = &Bsm[1][srow * ROWB + sc];

  const int rowA = m0 + r;
  const size_t srcRow = qremap
      ? (size_t)((rowA >> 10) * KLEN_ + MLEN_ + (rowA & (QLEN_ - 1)))
      : (size_t)rowA;
  const bf16_t* arow = A + srcRow * C_;

  async_ld_b128(gsrc, l0);                     // stage k0 = 0
  async_ld_b128(gsrc + 8, l0 + 8);
  v16bf aCur = load_frag_bf(arow, hs);
  v8f acc[8] = {};
  int cur = 0;
  for (int k0 = 0; k0 < C_; k0 += 32) {
    wait_async0();
    __syncthreads();
    const bool more = (k0 + 32 < C_);
    if (more) {
      const bf16_t* g = gsrc + k0 + 32;
      bf16_t* l = cur ? l0 : l1;
      async_ld_b128(g, l);
      async_ld_b128(g + 8, l + 8);
    }
    v16bf aNext = more ? load_frag_bf(arow + k0 + 32, hs) : aCur;
    const bf16_t* bbuf = cur ? &Bsm[1][0] : &Bsm[0][0];
    v16bf bfrag[8];
#pragma unroll
    for (int s = 0; s < 8; ++s)
      bfrag[s] = load_frag_bf(bbuf + (s * 16 + r) * ROWB, hs);
#pragma unroll
    for (int s = 0; s < 8; ++s)
      acc[s] = wmma_bf16(aCur, bfrag[s], acc[s]);
    aCur = aNext;
    cur ^= 1;
  }

  const int mb = m0 + hs * 8;
#pragma unroll
  for (int s = 0; s < 8; ++s) {
    const int col = n0 + s * 16 + r;
    const float b0 = bias0 ? bias0[col] : 0.f;
    const float b1 = bias1 ? bias1[col] : 0.f;
#pragma unroll
    for (int vg = 0; vg < 8; ++vg) {
      const int row = mb + vg;
      const float val = acc[s][vg];
      if (vtrans) {
        const int bb = row >> 11, j = row & (KLEN_ - 1);
        const int hh = col >> 6,  d = col & 63;
        out0[((size_t)(bb * H_ + hh) * DK_ + d) * KLEN_ + j] = (bf16_t)val;
      } else {
        out0[(size_t)row * C_ + col] = (bf16_t)(val + b0);
        if (out1) out1[(size_t)row * C_ + col] = (bf16_t)(val + b1);
      }
    }
  }
}

// ---- score GEMM: per (b,h): OUT[i,j] = sum_d Aq[b,i,h*64+d] * Bk[j,h*64+d]
// K = 64 only (2 WMMAs); store-bandwidth-bound (writes 256 MB fp32), so no
// LDS staging. OUT in (b,h,i,j) layout (= transposed aw the reference returns).
__global__ void __launch_bounds__(256)
score_gemm(const bf16_t* __restrict__ Aq, const bf16_t* __restrict__ Bk,
           int bBatched, float* __restrict__ out)
{
  const int lane = threadIdx.x & 31;
  const int gw   = blockIdx.x * (blockDim.x >> 5) + (threadIdx.x >> 5);
  const int bh  = gw >> 11;           // per (b,h): 64 m-tiles * 32 n-groups
  const int t   = gw & 2047;
  const int tm  = t >> 5;
  const int tng = t & 31;
  const int b = bh >> 4, h = bh & 15;
  const int m0 = tm * 16, n0 = tng * 64;
  const int r  = lane & 15, hs = (lane >> 4) & 1;

  const bf16_t* arow = Aq + ((size_t)b * QLEN_ + m0 + r) * C_ + h * DK_;
  const v16bf a0 = load_frag_bf(arow, hs);
  const v16bf a1 = load_frag_bf(arow + 32, hs);

  const size_t bRowBase = bBatched ? (size_t)b * KLEN_ : 0;
  float* obase = out + (size_t)bh * QLEN_ * KLEN_;

#pragma unroll
  for (int s = 0; s < 4; ++s) {
    const int n = n0 + s * 16;
    const bf16_t* brow = Bk + (bRowBase + (size_t)(n + r)) * C_ + h * DK_;
    v8f acc = {};
    acc = wmma_bf16(a0, load_frag_bf(brow, hs), acc);
    acc = wmma_bf16(a1, load_frag_bf(brow + 32, hs), acc);
    const int col = n + r;
    const int mb  = m0 + hs * 8;
#pragma unroll
    for (int vg = 0; vg < 8; ++vg)
      obase[(size_t)(mb + vg) * KLEN_ + col] = acc[vg];
  }
}

// ---- rel-shift gather + mask + scale + row softmax (in place in aw) -------
// _rel_shift exact: out[i,j] = xp[n/(K+1), n%(K+1)], n = i*K + j + Q,
// xp[:,0] = 0, xp[:,j'] = BD[:, j'-1].
__global__ void __launch_bounds__(256)
bd_softmax(float* __restrict__ aw, const float* __restrict__ bdf,
           const int* __restrict__ mask)
{
  const int row = blockIdx.x;            // over B*H*QLEN
  const int bh  = row >> 10;
  const int i2  = row & (QLEN_ - 1);
  const int b   = bh >> 4;
  const int tid = threadIdx.x;
  const int lane = tid & 31, wid = tid >> 5;

  float* arow = aw + (size_t)row * KLEN_;
  const float* bdbase = bdf + (size_t)bh * QLEN_ * KLEN_;
  const int* mrow = mask + ((size_t)b * QLEN_ + i2) * KLEN_;
  const float NEG = -3.4028234663852886e38f;

  float e[8];
#pragma unroll
  for (int l = 0; l < 8; ++l) {
    const int j = tid + l * 256;
    const int n = i2 * KLEN_ + j + QLEN_;
    const int isrc = n / (KLEN_ + 1);
    const int jp   = n % (KLEN_ + 1);
    const float bd = (jp >= 1) ? bdbase[(size_t)isrc * KLEN_ + (jp - 1)] : 0.f;
    const float v  = (arow[j] + bd) * 0.125f;     // 1/sqrt(64)
    e[l] = (mrow[j] == 0) ? NEG : v;
  }

  __shared__ float red[8];
  float m = e[0];
#pragma unroll
  for (int l = 1; l < 8; ++l) m = fmaxf(m, e[l]);
#pragma unroll
  for (int o = 16; o > 0; o >>= 1) m = fmaxf(m, __shfl_xor(m, o, 32));
  if (lane == 0) red[wid] = m;
  __syncthreads();
  float gm = red[0];
#pragma unroll
  for (int w = 1; w < 8; ++w) gm = fmaxf(gm, red[w]);
  __syncthreads();

  float s = 0.f;
#pragma unroll
  for (int l = 0; l < 8; ++l) { e[l] = __expf(e[l] - gm); s += e[l]; }
#pragma unroll
  for (int o = 16; o > 0; o >>= 1) s += __shfl_xor(s, o, 32);
  if (lane == 0) red[wid] = s;
  __syncthreads();
  float gs = red[0];
#pragma unroll
  for (int w = 1; w < 8; ++w) gs += red[w];
  const float inv = 1.f / gs;

#pragma unroll
  for (int l = 0; l < 8; ++l) arow[tid + l * 256] = e[l] * inv;
}

// ---- cv GEMM: per (b,h): cv[i,d] = sum_j aw[b,h,i,j] * vT[b,h,d,j] --------
// Block: 8 waves = 128 q-rows x all 64 d-cols; vT k-slice staged in LDS
// (double-buffered async). blockIdx.x = bh * 8 + bq; grid exact.
__global__ void __launch_bounds__(256)
cv_gemm(const float* __restrict__ aw, const bf16_t* __restrict__ vT,
        bf16_t* __restrict__ cvb)
{
  __shared__ bf16_t Bsm[2][64 * ROWB];
  const int tid  = threadIdx.x;
  const int lane = tid & 31;
  const int wv   = tid >> 5;
  const int bh = blockIdx.x >> 3;
  const int bq = blockIdx.x & 7;
  const int b = bh >> 4, h = bh & 15;
  const int m0 = bq * 128 + wv * 16;
  const int r  = lane & 15, hs = (lane >> 4) & 1;

  const float*  arow  = aw + ((size_t)bh * QLEN_ + m0 + r) * KLEN_;
  const bf16_t* Bbase = vT + (size_t)bh * DK_ * KLEN_;

  const int srow = tid >> 2, schunk = tid & 3;
  const bf16_t* gsrc = Bbase + (size_t)srow * KLEN_ + schunk * 8;
  bf16_t* l0 = &Bsm[0][srow * ROWB + schunk * 8];
  bf16_t* l1 = &Bsm[1][srow * ROWB + schunk * 8];

  async_ld_b128(gsrc, l0);
  v16bf aCur = load_frag_f32(arow, hs);
  v8f acc[4] = {};
  int cur = 0;
  for (int k0 = 0; k0 < KLEN_; k0 += 32) {
    wait_async0();
    __syncthreads();
    const bool more = (k0 + 32 < KLEN_);
    if (more) async_ld_b128(gsrc + k0 + 32, cur ? l0 : l1);
    v16bf aNext = more ? load_frag_f32(arow + k0 + 32, hs) : aCur;
    const bf16_t* bbuf = cur ? &Bsm[1][0] : &Bsm[0][0];
    v16bf bfrag[4];
#pragma unroll
    for (int s = 0; s < 4; ++s)
      bfrag[s] = load_frag_bf(bbuf + (s * 16 + r) * ROWB, hs);
#pragma unroll
    for (int s = 0; s < 4; ++s)
      acc[s] = wmma_bf16(aCur, bfrag[s], acc[s]);
    aCur = aNext;
    cur ^= 1;
  }

  const int mb = m0 + hs * 8;
#pragma unroll
  for (int s = 0; s < 4; ++s) {
    const int d = s * 16 + r;
#pragma unroll
    for (int vg = 0; vg < 8; ++vg) {
      const int i = mb + vg;
      cvb[((size_t)b * QLEN_ + i) * C_ + h * DK_ + d] = (bf16_t)acc[s][vg];
    }
  }
}

// ---- final GEMM: OUT[M x C] = cvb[M x C] @ Wo^T, fp32 output --------------
// Same 128x128 cooperative LDS-staged structure as proj_gemm.
__global__ void __launch_bounds__(256)
out_gemm(const bf16_t* __restrict__ A, const bf16_t* __restrict__ W,
         float* __restrict__ out)
{
  __shared__ bf16_t Bsm[2][128 * ROWB];
  const int tid  = threadIdx.x;
  const int lane = tid & 31;
  const int wv   = tid >> 5;
  const int bm   = blockIdx.x >> 3;
  const int bn   = blockIdx.x & 7;
  const int m0   = bm * 128 + wv * 16;
  const int n0   = bn * 128;
  const int r    = lane & 15, hs = (lane >> 4) & 1;

  const int srow = tid >> 1, sc = (tid & 1) * 16;
  const bf16_t* gsrc = W + (size_t)(n0 + srow) * C_ + sc;
  bf16_t* l0 = &Bsm[0][srow * ROWB + sc];
  bf16_t* l1 = &Bsm[1][srow * ROWB + sc];

  const bf16_t* arow = A + (size_t)(m0 + r) * C_;

  async_ld_b128(gsrc, l0);
  async_ld_b128(gsrc + 8, l0 + 8);
  v16bf aCur = load_frag_bf(arow, hs);
  v8f acc[8] = {};
  int cur = 0;
  for (int k0 = 0; k0 < C_; k0 += 32) {
    wait_async0();
    __syncthreads();
    const bool more = (k0 + 32 < C_);
    if (more) {
      const bf16_t* g = gsrc + k0 + 32;
      bf16_t* l = cur ? l0 : l1;
      async_ld_b128(g, l);
      async_ld_b128(g + 8, l + 8);
    }
    v16bf aNext = more ? load_frag_bf(arow + k0 + 32, hs) : aCur;
    const bf16_t* bbuf = cur ? &Bsm[1][0] : &Bsm[0][0];
    v16bf bfrag[8];
#pragma unroll
    for (int s = 0; s < 8; ++s)
      bfrag[s] = load_frag_bf(bbuf + (s * 16 + r) * ROWB, hs);
#pragma unroll
    for (int s = 0; s < 8; ++s)
      acc[s] = wmma_bf16(aCur, bfrag[s], acc[s]);
    aCur = aNext;
    cur ^= 1;
  }

  const int mb = m0 + hs * 8;
#pragma unroll
  for (int s = 0; s < 8; ++s) {
    const int col = n0 + s * 16 + r;
#pragma unroll
    for (int vg = 0; vg < 8; ++vg)
      out[(size_t)(mb + vg) * C_ + col] = acc[s][vg];
  }
}

// ---------------------------------------------------------------------------
extern "C" void kernel_launch(void* const* d_in, const int* in_sizes, int n_in,
                              void* d_out, int out_size, void* d_ws, size_t ws_size,
                              hipStream_t stream) {
  (void)in_sizes; (void)n_in; (void)out_size; (void)ws_size;

  const float* key    = (const float*)d_in[0];
  // d_in[1] "query" is unused by the reference (it projects key[:, -Q:]).
  const float* pos    = (const float*)d_in[2];
  const int*   mask   = (const int*)d_in[3];
  const float* u_bias = (const float*)d_in[4];
  const float* v_bias = (const float*)d_in[5];
  const float* Wk     = (const float*)d_in[6];
  const float* Wv     = (const float*)d_in[7];
  const float* Wq     = (const float*)d_in[8];
  const float* Wo     = (const float*)d_in[9];
  const float* Wp     = (const float*)d_in[10];

  float* outp  = (float*)d_out;                       // cv: B*QLEN*C fp32
  float* awPtr = outp + (size_t)B_ * QLEN_ * C_;      // aw: B*H*QLEN*KLEN fp32

  char* ws = (char*)d_ws;
  size_t off = 0;
  auto take = [&](size_t bytes) -> void* {
    void* p = ws + off;
    off += (bytes + 255) & ~(size_t)255;
    return p;
  };
  const size_t NKC = (size_t)B_ * KLEN_ * C_;
  const size_t NQC = (size_t)B_ * QLEN_ * C_;
  const size_t NPC = (size_t)KLEN_ * C_;
  const size_t NW  = (size_t)C_ * C_;

  bf16_t* keyb = (bf16_t*)take(NKC * 2);
  bf16_t* posb = (bf16_t*)take(NPC * 2);
  bf16_t* wkb  = (bf16_t*)take(NW * 2);
  bf16_t* wvb  = (bf16_t*)take(NW * 2);
  bf16_t* wqb  = (bf16_t*)take(NW * 2);
  bf16_t* wpb  = (bf16_t*)take(NW * 2);
  bf16_t* wob  = (bf16_t*)take(NW * 2);
  bf16_t* kb   = (bf16_t*)take(NKC * 2);
  bf16_t* vTb  = (bf16_t*)take(NKC * 2);        // [B,H,DK,KLEN]
  bf16_t* qu   = (bf16_t*)take(NQC * 2);
  bf16_t* qv   = (bf16_t*)take(NQC * 2);
  bf16_t* pb   = (bf16_t*)take(NPC * 2);
  bf16_t* cvb  = (bf16_t*)take(NQC * 2);
  float*  bdf  = (float*)take((size_t)B_ * H_ * QLEN_ * KLEN_ * 4);  // 256 MB

  // 1) fp32 -> bf16 conversions
  cvt_bf16<<<1024, 256, 0, stream>>>(key, keyb, (int)NKC);
  cvt_bf16<<<512,  256, 0, stream>>>(pos, posb, (int)NPC);
  cvt_bf16<<<256,  256, 0, stream>>>(Wk, wkb, (int)NW);
  cvt_bf16<<<256,  256, 0, stream>>>(Wv, wvb, (int)NW);
  cvt_bf16<<<256,  256, 0, stream>>>(Wq, wqb, (int)NW);
  cvt_bf16<<<256,  256, 0, stream>>>(Wp, wpb, (int)NW);
  cvt_bf16<<<256,  256, 0, stream>>>(Wo, wob, (int)NW);

  // 2) projections (WMMA bf16, 128x128 tiles, LDS async-staged weights)
  proj_gemm<<<(B_ * KLEN_ / 128) * 8, 256, 0, stream>>>(
      keyb, wkb, kb,  nullptr, nullptr, nullptr, 0, 0);
  proj_gemm<<<(B_ * KLEN_ / 128) * 8, 256, 0, stream>>>(
      keyb, wvb, vTb, nullptr, nullptr, nullptr, 0, 1);
  proj_gemm<<<(B_ * QLEN_ / 128) * 8, 256, 0, stream>>>(
      keyb, wqb, qu,  qv, u_bias, v_bias, 1, 0);
  proj_gemm<<<(KLEN_ / 128) * 8, 256, 0, stream>>>(
      posb, wpb, pb,  nullptr, nullptr, nullptr, 0, 0);

  // 3) scores: AC into aw region of d_out, BD into workspace
  score_gemm<<<8192, 256, 0, stream>>>(qu, kb, 1, awPtr);
  score_gemm<<<8192, 256, 0, stream>>>(qv, pb, 0, bdf);

  // 4) rel-shift + mask + scale + softmax (in place; aw now final output)
  bd_softmax<<<B_ * H_ * QLEN_, 256, 0, stream>>>(awPtr, bdf, mask);

  // 5) context vectors and output projection
  cv_gemm<<<B_ * H_ * (QLEN_ / 128), 256, 0, stream>>>(awPtr, vTb, cvb);
  out_gemm<<<(B_ * QLEN_ / 128) * 8, 256, 0, stream>>>(cvb, wob, outp);
}
